// ExpertPreferredRouter_11854109737664
// MI455X (gfx1250) — matrix operand order repaired and checked by reading, not compiled, not run on hardware
//
#include <hip/hip_runtime.h>

typedef __attribute__((ext_vector_type(16))) __bf16 v16bf;
typedef __attribute__((ext_vector_type(8)))  __bf16 v8bf;
typedef __attribute__((ext_vector_type(8)))  float  v8f;
typedef __attribute__((ext_vector_type(4)))  int    v4i;
typedef __attribute__((ext_vector_type(4)))  unsigned int uv4;
typedef __attribute__((ext_vector_type(8)))  int    iv8;
typedef __attribute__((ext_vector_type(4)))  int    iv4;

#define B_ 4
#define N_ 4096
#define D_ 4096
#define E_ 64

#define XS_STRIDE 36   // 128-row fp32 x tile, padded: banks 36*r mod 64 all distinct
#define WS_STRIDE 40   // 64-row bf16 W tiles, padded (80B rows, 16B aligned)
#define LG_STRIDE 65   // logits tile, padded so per-thread row walks rotate banks

#if __has_builtin(__builtin_amdgcn_global_load_async_to_lds_b128) && \
    __has_builtin(__builtin_amdgcn_s_wait_asynccnt)
#define USE_ASYNC 1
#else
#define USE_ASYNC 0
#endif

#if __has_builtin(__builtin_amdgcn_tensor_load_to_lds) && \
    __has_builtin(__builtin_amdgcn_s_wait_tensorcnt)
#define USE_TDM 1
#else
#define USE_TDM 0
#endif

// async-group size per wave per k-step (W tiles only when TDM moves x)
#define AW_GROUP (USE_TDM ? 2 : 6)

// copy 16 bytes global -> LDS (async when available)
__device__ __forceinline__ void cp16(void* lds, const void* g) {
#if USE_ASYNC
  __builtin_amdgcn_global_load_async_to_lds_b128(
      (__attribute__((address_space(1))) v4i*)g,
      (__attribute__((address_space(3))) v4i*)lds, 0, 0);
#else
  *(float4*)lds = *(const float4*)g;
#endif
}

#if USE_TDM
// TDM: move a 128x32 fp32 tile (row stride D_) into LDS with 4-DWORD padding
// after every 32-DWORD row -> LDS row stride = 36 floats (XS_STRIDE).
__device__ __forceinline__ void tdm_load_x(const void* gsrc, unsigned lds_off) {
  const unsigned long long ga = (unsigned long long)(size_t)gsrc;
  uv4 g0 = {
      1u,                                   // count=1, load, user mode
      lds_off,                              // LDS byte address
      (unsigned)(ga & 0xFFFFFFFFu),         // global_addr[31:0]
      (unsigned)((ga >> 32) & 0x1FFFFFFu)   // global_addr[56:32]
          | (2u << 30)                      // type = 2 (image)
  };
  iv8 g1 = {
      (2 << 16) | (1 << 20) | (4 << 22) | (3 << 25), // 4B elems; pad_en; every 32 DW pad 4 DW
      (int)((D_ & 0xFFFF) << 16),                    // tensor_dim0 lo16 (4096)
      (int)(((unsigned)D_ >> 16) | ((unsigned)(B_ * N_ & 0xFFFF) << 16)), // dim0 hi | dim1 lo
      (int)(32u << 16),                              // dim1 hi | tile_dim0 = 32
      128,                                           // tile_dim1 = 128 | tile_dim2 = 0
      D_,                                            // tensor_dim0_stride lo (4096)
      0, 0
  };
  iv4 z4 = {0, 0, 0, 0};
  iv8 z8 = {0, 0, 0, 0, 0, 0, 0, 0};
  // 6-arg toolchain variant: (g0, g1, g2, g3, extra, cpol)
  __builtin_amdgcn_tensor_load_to_lds(g0, g1, z4, z4, z8, 0);
}
#endif

// ---------------------------------------------------------------------------
// Prep: split W (fp32) into bf16 hi/lo so the GEMM's B operand needs no
// per-k-step conversion (W is tiny: 64x4096).
// ---------------------------------------------------------------------------
__global__ __launch_bounds__(256) void split_w(const float* __restrict__ W,
                                               __bf16* __restrict__ whi,
                                               __bf16* __restrict__ wlo) {
  const int i = blockIdx.x * 256 + threadIdx.x;
  const float f = W[i];
  const __bf16 h = (__bf16)f;
  whi[i] = h;
  wlo[i] = (__bf16)(f - (float)h);
}

// ---------------------------------------------------------------------------
// Kernel 1: router GEMM (bf16x3 split precision, v_wmma_f32_16x16x32_bf16),
// x tile staged by the Tensor Data Mover, W tiles by async-to-LDS loads,
// double buffered; fused softmax over E=64.
// grid = (B*N)/128, 256 threads = 8 wave32s.
// ---------------------------------------------------------------------------
__global__ __launch_bounds__(256) void router_gemm_softmax(
    const float* __restrict__ x,
    const __bf16* __restrict__ whiG, const __bf16* __restrict__ wloG,
    float* __restrict__ probsT)
{
  __shared__ __align__(16) float  xs[2 * 128 * XS_STRIDE];  // 36 KB
  __shared__ __align__(16) __bf16 wh[2 * E_ * WS_STRIDE];   // 10 KB
  __shared__ __align__(16) __bf16 wl[2 * E_ * WS_STRIDE];   // 10 KB
  __shared__ float lg[128 * LG_STRIDE];                     // ~33 KB

  const int  tid  = threadIdx.x;
  const int  lane = tid & 31;
  const int  wave = tid >> 5;
  const long rowBlock = (long)blockIdx.x * 128;

  // staging geometry (per thread: 1x16B of Whi, 1x16B of Wlo; x via TDM)
  const int sr = tid >> 1;              // x row 0..127 (fallback path)
  const int sh = (tid & 1) << 4;        // x col half: 0 or 16 floats
  const int se = tid >> 2;              // W row 0..63
  const int sq = (tid & 3) << 3;        // W col chunk: 0/8/16/24 bf16

  auto stage = [&](int kb, int buf) {
#if USE_TDM
    if (tid < 32)   // wave 0 issues one TDM op; EXEC-independent DMA
      tdm_load_x(x + rowBlock * (long)D_ + kb,
                 (unsigned)(size_t)(void*)&xs[buf * 128 * XS_STRIDE]);
#else
    const float* gx = x + (rowBlock + sr) * (long)D_ + kb + sh;
    float* lx = &xs[buf * 128 * XS_STRIDE + sr * XS_STRIDE + sh];
#pragma unroll
    for (int i = 0; i < 4; ++i) cp16(lx + 4 * i, gx + 4 * i);
#endif
    cp16(&wh[buf * E_ * WS_STRIDE + se * WS_STRIDE + sq], whiG + (long)se * D_ + kb + sq);
    cp16(&wl[buf * E_ * WS_STRIDE + se * WS_STRIDE + sq], wloG + (long)se * D_ + kb + sq);
  };

  v8f acc[4] = {v8f{}, v8f{}, v8f{}, v8f{}};

  // fragment geometry
  const int rloc = (wave << 4) + (lane & 15);   // A row within 128-row tile
  const int k0   = (lane < 16) ? 0 : 8;         // A: K={0..7,16..23} / {8..15,24..31}
  const int koff = (lane < 16) ? 0 : 8;         // B: K={0..7,16..23} / {8..15,24..31}
  const int cl   = lane & 15;

  stage(0, 0);
  const int KSTEPS = D_ / 32;
  for (int kbi = 0; kbi < KSTEPS; ++kbi) {
    const int buf = kbi & 1;
    if (kbi + 1 < KSTEPS) {
      stage((kbi + 1) << 5, buf ^ 1);
#if USE_ASYNC
      __builtin_amdgcn_s_wait_asynccnt(AW_GROUP);  // previous W group done (in-order)
#endif
#if USE_TDM
      if (tid < 32) __builtin_amdgcn_s_wait_tensorcnt(1);  // previous x tile done
#endif
    } else {
#if USE_ASYNC
      __builtin_amdgcn_s_wait_asynccnt(0);
#endif
#if USE_TDM
      if (tid < 32) __builtin_amdgcn_s_wait_tensorcnt(0);
#endif
    }
    __syncthreads();

    const float*  xsB = &xs[buf * 128 * XS_STRIDE];
    const __bf16* whB = &wh[buf * E_ * WS_STRIDE];
    const __bf16* wlB = &wl[buf * E_ * WS_STRIDE];

    // ---- A fragment: fp32 -> bf16 hi/lo split (only conversion left) ----
    v16bf ah, al;
#pragma unroll
    for (int i = 0; i < 8; ++i) {
      float  f = xsB[rloc * XS_STRIDE + k0 + i];
      __bf16 h = (__bf16)f;
      ah[i] = h; al[i] = (__bf16)(f - (float)h);
      f = xsB[rloc * XS_STRIDE + k0 + 16 + i];
      h = (__bf16)f;
      ah[8 + i] = h; al[8 + i] = (__bf16)(f - (float)h);
    }
#pragma unroll
    for (int c = 0; c < 4; ++c) {
      const int e = (c << 4) + cl;
      const __bf16* ph = &whB[e * WS_STRIDE + koff];
      const __bf16* pl = &wlB[e * WS_STRIDE + koff];
      v8bf h0 = *(const v8bf*)ph, h1 = *(const v8bf*)(ph + 16);
      v8bf l0 = *(const v8bf*)pl, l1 = *(const v8bf*)(pl + 16);
      v16bf bh = __builtin_shufflevector(h0, h1, 0,1,2,3,4,5,6,7,8,9,10,11,12,13,14,15);
      v16bf bl = __builtin_shufflevector(l0, l1, 0,1,2,3,4,5,6,7,8,9,10,11,12,13,14,15);
      // bf16x3: hi*hi + hi*lo + lo*hi  (lo*lo below fp32 ulp)
      acc[c] = __builtin_amdgcn_wmma_f32_16x16x32_bf16(false, ah, false, bh,
                                                       (short)0, acc[c], false, false);
      acc[c] = __builtin_amdgcn_wmma_f32_16x16x32_bf16(false, ah, false, bl,
                                                       (short)0, acc[c], false, false);
      acc[c] = __builtin_amdgcn_wmma_f32_16x16x32_bf16(false, al, false, bh,
                                                       (short)0, acc[c], false, false);
    }
    __syncthreads();   // protect buf before it is refilled next-next step
  }

  // ---- dump D tiles: row = base + v + 8*(lane/16), col = lane%16 ----
  const int rb = (wave << 4) + ((lane >> 4) << 3);
#pragma unroll
  for (int c = 0; c < 4; ++c)
#pragma unroll
    for (int v = 0; v < 8; ++v)
      lg[(rb + v) * LG_STRIDE + (c << 4) + cl] = acc[c][v];
  __syncthreads();

  // ---- softmax per row (128 rows, one thread each); coalesced probsT writes ----
  if (tid < 128) {
    const float* row = &lg[tid * LG_STRIDE];
    float m = row[0];
    for (int e = 1; e < E_; ++e) m = fmaxf(m, row[e]);
    float s = 0.f;
    for (int e = 0; e < E_; ++e) s += __expf(row[e] - m);
    const float inv = 1.f / s;
    const long g = rowBlock + tid;
    const long b = g / N_;
    const long n = g % N_;
    float* dst = probsT + b * (long)E_ * N_ + n;
    for (int e = 0; e < E_; ++e)
      dst[(long)e * N_] = __expf(row[e] - m) * inv;   // threads write consecutive n
  }
}

// ---------------------------------------------------------------------------
// Kernel 2: greedy expert-preferred routing. One block per batch (B=4),
// 1024 threads. For j = E-1..0: exact top-K (K = floor(c[j]*N)) over unmasked
// tokens via MSB-first 8-bit radix select on float-bit keys, ties broken at
// the smallest index (stable like lax.top_k), then mask winners.
// ---------------------------------------------------------------------------
__global__ __launch_bounds__(1024) void expert_greedy_route(
    const float* __restrict__ probsT, const float* __restrict__ cap,
    float* __restrict__ outM, float* __restrict__ outP)
{
  __shared__ unsigned      key[N_];    // 16 KB: float bits + 1 (0 == masked)
  __shared__ int           hist[256];
  __shared__ int           sfx[256];
  __shared__ unsigned char msk[N_];    // 4 KB
  __shared__ int           shv[4];     // [0]=boundary bin, [1]=count above, [2]=tie idx

  const int tid = threadIdx.x;
  const int b   = blockIdx.x;
  const float* Pb   = probsT + (long)b * E_ * N_;
  float*       Mrow = outM + (long)b * N_;
  float*       Prow = outP + (long)b * N_;

  for (int t = tid; t < N_; t += 1024) {   // defaults: M = 0, M_probs = probs[:,:,0]
    msk[t]  = 0;
    Mrow[t] = 0.0f;
    Prow[t] = Pb[t];
  }
  __syncthreads();

  for (int j = E_ - 1; j >= 0; --j) {
    const int K = (int)floorf(cap[j] * (float)N_);
    if (K <= 0) continue;
    const float* row = Pb + (long)j * N_;

    for (int t = tid; t < N_; t += 1024)
      key[t] = msk[t] ? 0u : (__float_as_uint(row[t]) + 1u);  // probs >= 0
    __syncthreads();

    unsigned pref = 0u;
    int need = K;
#pragma unroll
    for (int lvl = 0; lvl < 4; ++lvl) {           // MSB-first 8-bit radix select
      const int s = 24 - (lvl << 3);
      if (tid < 256) hist[tid] = 0;
      if (tid == 0) { shv[0] = 0; shv[1] = 0; }
      __syncthreads();
      for (int t = tid; t < N_; t += 1024) {
        const unsigned k = key[t];
        if (k != 0u && (lvl == 0 || (k >> ((s + 8) & 31)) == pref))
          atomicAdd(&hist[(k >> s) & 0xFF], 1);
      }
      __syncthreads();
      if (tid < 256) sfx[tid] = hist[tid];
      __syncthreads();
      for (int off = 1; off < 256; off <<= 1) {   // Hillis-Steele suffix sum
        int v = 0;
        if (tid < 256) {
          v = sfx[tid];
          if (tid + off < 256) v += sfx[tid + off];
        }
        __syncthreads();
        if (tid < 256) sfx[tid] = v;
        __syncthreads();
      }
      if (tid < 256) {
        const int above = (tid == 255) ? 0 : sfx[tid + 1];
        if (sfx[tid] >= need && above < need) { shv[0] = tid; shv[1] = above; }
      }
      __syncthreads();
      pref  = (pref << 8) | (unsigned)shv[0];
      need -= shv[1];
      __syncthreads();
    }
    const unsigned Kb = pref;   // exact boundary key; take `need` ties

    for (int t = tid; t < N_; t += 1024) {        // strictly above boundary
      if (key[t] > Kb) {
        msk[t]  = 1;
        Mrow[t] = (float)j;
        Prow[t] = __uint_as_float(key[t] - 1u);
      }
    }
    int last = -1;                                 // ties: smallest indices
    for (int r = 0; r < need; ++r) {
      if (tid == 0) shv[2] = N_;
      __syncthreads();
      for (int t = tid; t < N_; t += 1024)
        if (key[t] == Kb && t > last) atomicMin(&shv[2], t);
      __syncthreads();
      const int pick = shv[2];
      if (pick >= N_) break;                       // uniform: shv[2] shared
      if (tid == 0) {
        msk[pick]  = 1;
        Mrow[pick] = (float)j;
        Prow[pick] = __uint_as_float(Kb - 1u);
      }
      last = pick;
      __syncthreads();
    }
    __syncthreads();   // msk visible before next expert's key build
  }
}

extern "C" void kernel_launch(void* const* d_in, const int* in_sizes, int n_in,
                              void* d_out, int out_size, void* d_ws, size_t ws_size,
                              hipStream_t stream) {
  const float* x = (const float*)d_in[0];   // [B,N,D] fp32
  const float* W = (const float*)d_in[1];   // [E,D]   fp32
  const float* c = (const float*)d_in[2];   // [E]     fp32

  char* ws = (char*)d_ws;
  float*  probsT = (float*)ws;                                   // [B,E,N] 4 MiB
  __bf16* whi    = (__bf16*)(ws + (size_t)B_ * E_ * N_ * 4);     // [E,D] 512 KiB
  __bf16* wlo    = whi + (size_t)E_ * D_;                        // [E,D] 512 KiB

  float* outM = (float*)d_out;                   // [B,N] expert id (as float)
  float* outP = (float*)d_out + (long)B_ * N_;   // [B,N] chosen prob

  split_w<<<(E_ * D_) / 256, 256, 0, stream>>>(W, whi, wlo);
  router_gemm_softmax<<<(B_ * N_) / 128, 256, 0, stream>>>(x, whi, wlo, probsT);
  expert_greedy_route<<<B_, 1024, 0, stream>>>(probsT, c, outM, outP);
}
